// APSA_51049981281500
// MI455X (gfx1250) — compile-verified, matrix operand order checked by reading, not run on hardware
//
#include <hip/hip_runtime.h>
#include <stdint.h>

// ---------------- dims ----------------
#define EPS_  1e-5f
#define INV_SQRT_D 0.08838834764831845f   // 1/sqrt(128)

typedef __attribute__((ext_vector_type(16))) __bf16 v16bf;
typedef __attribute__((ext_vector_type(8)))  float  v8f;

__device__ __forceinline__ __bf16 f2bf(float f) {
    unsigned int u = __float_as_uint(f);
    unsigned int r = (u + 0x7FFFu + ((u >> 16) & 1u)) >> 16;
    unsigned short s = (unsigned short)r;
    return __builtin_bit_cast(__bf16, s);
}

__device__ __forceinline__ float block_reduce_sum(float v, float* red) {
    int tid = threadIdx.x;
    red[tid] = v; __syncthreads();
    #pragma unroll
    for (int s = 128; s > 0; s >>= 1) {
        if (tid < s) red[tid] += red[tid + s];
        __syncthreads();
    }
    float r = red[0]; __syncthreads();
    return r;
}

// ---- CDNA5 async global->LDS copy (16B per lane), tracked on ASYNCcnt ----
__device__ __forceinline__ void async_copy16(void* lds_generic, const void* gptr) {
    unsigned lds = (unsigned)(size_t)lds_generic;   // low 32 bits = LDS byte address
    asm volatile("global_load_async_to_lds_b128 %0, %1, off"
                 :: "v"(lds), "v"(gptr)
                 : "memory");
}
__device__ __forceinline__ void wait_async0() {
    asm volatile("s_wait_asynccnt 0x0" ::: "memory");
}

// ---------------- f32 -> bf16 transposed convert: in[K][N] -> out[N][K] ----------------
__global__ __launch_bounds__(256) void cvt_transpose_kernel(const float* __restrict__ in,
                                                            __bf16* __restrict__ out,
                                                            int K, int N) {
    __shared__ float tile[32][33];
    const int kb = blockIdx.y * 32, nb = blockIdx.x * 32;
    const int tx = threadIdx.x & 31, ty = threadIdx.x >> 5;   // 8 rows per pass
    #pragma unroll
    for (int r = 0; r < 32; r += 8)
        tile[ty + r][tx] = in[(size_t)(kb + ty + r) * N + nb + tx];
    __syncthreads();
    #pragma unroll
    for (int r = 0; r < 32; r += 8)
        out[(size_t)(nb + ty + r) * K + kb + tx] = f2bf(tile[tx][ty + r]);
}

// ---------------- bf16 WMMA GEMM: C[MxN] = A[MxK] @ Bt[NxK]^T ----------------
// 256 threads = 8 wave32 waves; 128x128 tile, K-step 32, double-buffered LDS fed by
// global_load_async_to_lds_b128.
__global__ __launch_bounds__(256) void gemm_bf16_kernel(const __bf16* __restrict__ A,
                                                        const __bf16* __restrict__ Bt,
                                                        float* __restrict__ C,
                                                        int M, int N, int K) {
    __shared__ __bf16 As[2][128][32];   // [m][k], k contiguous (64B rows)
    __shared__ __bf16 Bs[2][128][32];   // [n][k], k contiguous

    const int tid  = threadIdx.x;
    const int wave = tid >> 5;
    const int lane = tid & 31;
    const int lrow = lane & 15;   // M (A/C) or N (B) within fragment
    const int lhi  = lane >> 4;   // K-group / M-half select
    const int bm = blockIdx.y * 128;
    const int bn = blockIdx.x * 128;

    v8f acc[8];
    #pragma unroll
    for (int i = 0; i < 8; ++i)
        #pragma unroll
        for (int j = 0; j < 8; ++j) acc[i][j] = 0.0f;

    const int nk = K >> 5;

    auto issue_tile = [&](int buf, int kt) {
        const int k0 = kt << 5;
        #pragma unroll
        for (int i = 0; i < 2; ++i) {
            int idx = tid + i * 256;            // 0..511 vectors of 8 bf16
            int m = idx >> 2, g = (idx & 3) << 3;
            async_copy16(&As[buf][m][g], &A [(size_t)(bm + m) * K + k0 + g]);
            async_copy16(&Bs[buf][m][g], &Bt[(size_t)(bn + m) * K + k0 + g]);
        }
    };

    issue_tile(0, 0);
    wait_async0();
    __syncthreads();

    for (int kt = 0; kt < nk; ++kt) {
        const int cur = kt & 1;
        if (kt + 1 < nk) issue_tile(cur ^ 1, kt + 1);
        if (kt + 4 < nk)   // L2 prefetch (global_prefetch_b8) for the streaming operand
            __builtin_prefetch(&Bt[(size_t)bn * K + ((kt + 4) << 5)], 0, 1);

        // A fragment: rows (wave*16 + lrow); ISA 16-bit layout: two 16B runs per lane
        v16bf af;
        {
            const __bf16* ap = &As[cur][wave * 16 + lrow][lhi * 8];
            #pragma unroll
            for (int e = 0; e < 8; ++e) { af[e] = ap[e]; af[e + 8] = ap[16 + e]; }
        }
        // load ALL B fragments first, then burst the 8 WMMAs
        v16bf bfr[8];
        #pragma unroll
        for (int nf = 0; nf < 8; ++nf) {
            const __bf16* bp = &Bs[cur][nf * 16 + lrow][lhi * 8];
            #pragma unroll
            for (int e = 0; e < 8; ++e) { bfr[nf][e] = bp[e]; bfr[nf][e + 8] = bp[16 + e]; }
        }
        #pragma unroll
        for (int nf = 0; nf < 8; ++nf)
            acc[nf] = __builtin_amdgcn_wmma_f32_16x16x32_bf16(
                false, af, false, bfr[nf], (short)0, acc[nf], false, false);

        wait_async0();
        __syncthreads();
    }

    // C fragment layout: lane<16 -> M=r, lane>=16 -> M=8+r ; N=lrow
    #pragma unroll
    for (int nf = 0; nf < 8; ++nf)
        #pragma unroll
        for (int r = 0; r < 8; ++r)
            C[(size_t)(bm + wave * 16 + lhi * 8 + r) * N + bn + nf * 16 + lrow] = acc[nf][r];
}

// ---------------- patch gather + LayerNorm -> FLbf (bf16) ----------------
__global__ __launch_bounds__(256) void patch_ln_kernel(const float* __restrict__ x,
                                                       const float* __restrict__ g,
                                                       const float* __restrict__ bt,
                                                       __bf16* __restrict__ out) {
    const int t = blockIdx.x;                 // (b*16+win)*16+pat
    const int pat = t & 15, win = (t >> 4) & 15, b = t >> 8;
    const int wy = win >> 2, wx = win & 3, qy = pat >> 2, qx = pat & 3;
    float vals[4], s = 0.f, ss = 0.f;
    #pragma unroll
    for (int j = 0; j < 4; ++j) {
        int d = threadIdx.x + j * 256;
        int py = d >> 7, px = (d >> 4) & 7, c = d & 15;
        int Y = wy * 32 + qy * 8 + py, X = wx * 32 + qx * 8 + px;
        float v = x[(((size_t)b * 16 + c) * 128 + Y) * 128 + X];
        vals[j] = v; s += v; ss += v * v;
    }
    __shared__ float red[256];
    float sum = block_reduce_sum(s, red);
    float sq  = block_reduce_sum(ss, red);
    float mean = sum * (1.f / 1024.f);
    float rstd = rsqrtf(sq * (1.f / 1024.f) - mean * mean + EPS_);
    #pragma unroll
    for (int j = 0; j < 4; ++j) {
        int d = threadIdx.x + j * 256;
        out[(size_t)t * 1024 + d] = f2bf((vals[j] - mean) * rstd * g[d] + bt[d]);
    }
}

// ---------------- pooling (4x4 avg) + LayerNorm -> PLn (bf16) ----------------
__global__ __launch_bounds__(256) void pool_ln_kernel(const float* __restrict__ x,
                                                      const float* __restrict__ g,
                                                      const float* __restrict__ bt,
                                                      __bf16* __restrict__ out) {
    const int r = blockIdx.x;                 // b*16+win
    const int win = r & 15, b = r >> 4;
    const int wy = win >> 2, wx = win & 3;
    float vals[4], s = 0.f, ss = 0.f;
    #pragma unroll
    for (int j = 0; j < 4; ++j) {
        int d = threadIdx.x + j * 256;
        int iy = d >> 7, ix = (d >> 4) & 7, c = d & 15;
        int Y = (wy * 8 + iy) * 4, X = (wx * 8 + ix) * 4;
        float a = 0.f;
        #pragma unroll
        for (int dy = 0; dy < 4; ++dy)
            #pragma unroll
            for (int dx = 0; dx < 4; ++dx)
                a += x[(((size_t)b * 16 + c) * 128 + Y + dy) * 128 + X + dx];
        float v = a * (1.f / 16.f);
        vals[j] = v; s += v; ss += v * v;
    }
    __shared__ float red[256];
    float sum = block_reduce_sum(s, red);
    float sq  = block_reduce_sum(ss, red);
    float mean = sum * (1.f / 1024.f);
    float rstd = rsqrtf(sq * (1.f / 1024.f) - mean * mean + EPS_);
    #pragma unroll
    for (int j = 0; j < 4; ++j) {
        int d = threadIdx.x + j * 256;
        out[(size_t)r * 1024 + d] = f2bf((vals[j] - mean) * rstd * g[d] + bt[d]);
    }
}

// ---------------- per-window LayerNorm over WD=16384 -> XWn (bf16) ----------------
__global__ __launch_bounds__(256) void xw_ln_kernel(const float* __restrict__ x,
                                                    const float* __restrict__ g,
                                                    const float* __restrict__ bt,
                                                    __bf16* __restrict__ out) {
    const int r = blockIdx.x;                 // b*16+win
    const int win = r & 15, b = r >> 4;
    const int wy = win >> 2, wx = win & 3;
    float s = 0.f, ss = 0.f;
    for (int j = 0; j < 64; ++j) {
        int d = threadIdx.x + j * 256;
        int sy = d >> 9, sx = (d >> 4) & 31, c = d & 15;
        float v = x[(((size_t)b * 16 + c) * 128 + wy * 32 + sy) * 128 + wx * 32 + sx];
        s += v; ss += v * v;
    }
    __shared__ float red[256];
    float sum = block_reduce_sum(s, red);
    float sq  = block_reduce_sum(ss, red);
    float mean = sum * (1.f / 16384.f);
    float rstd = rsqrtf(sq * (1.f / 16384.f) - mean * mean + EPS_);
    for (int j = 0; j < 64; ++j) {
        int d = threadIdx.x + j * 256;
        int sy = d >> 9, sx = (d >> 4) & 31, c = d & 15;
        float v = x[(((size_t)b * 16 + c) * 128 + wy * 32 + sy) * 128 + wx * 32 + sx];
        out[(size_t)r * 16384 + d] = f2bf((v - mean) * rstd * g[d] + bt[d]);
    }
}

// ---------------- patch attention (16x16 per head) + LePE conv -> PFm (bf16) ----------------
__global__ __launch_bounds__(256) void patch_attn_kernel(const float* __restrict__ QK,
                                                         const float* __restrict__ PV,
                                                         const float* __restrict__ cw,
                                                         const float* __restrict__ cb,
                                                         __bf16* __restrict__ out) {
    const int blk = blockIdx.x;               // (b*16+win)*8 + h
    const int h = blk & 7;
    const size_t trow0 = (size_t)(blk >> 3) * 16;
    const int tid = threadIdx.x;

    __shared__ float sc[16][16];
    __shared__ float pvs[16][128];

    #pragma unroll
    for (int j = 0; j < 8; ++j) {
        int idx = tid + j * 256;
        int vi = idx >> 7, d = idx & 127;
        pvs[vi][d] = PV[(trow0 + vi) * 1024 + h * 128 + d];
    }
    const int qi = tid >> 4, ki = tid & 15;
    {
        const float4* qp = (const float4*)&QK[(trow0 + qi) * 2048 + h * 128];
        const float4* kp = (const float4*)&QK[(trow0 + ki) * 2048 + 1024 + h * 128];
        float dot = 0.f;
        #pragma unroll 8
        for (int d4 = 0; d4 < 32; ++d4) {
            float4 a = qp[d4], b = kp[d4];
            dot += a.x * b.x + a.y * b.y + a.z * b.z + a.w * b.w;
        }
        sc[qi][ki] = dot * INV_SQRT_D;
    }
    __syncthreads();
    if (tid < 16) {
        float mx = -1e30f;
        #pragma unroll
        for (int k = 0; k < 16; ++k) mx = fmaxf(mx, sc[tid][k]);
        float sm = 0.f;
        #pragma unroll
        for (int k = 0; k < 16; ++k) { float e = __expf(sc[tid][k] - mx); sc[tid][k] = e; sm += e; }
        float inv = 1.f / sm;
        #pragma unroll
        for (int k = 0; k < 16; ++k) sc[tid][k] *= inv;
    }
    __syncthreads();
    #pragma unroll
    for (int j = 0; j < 8; ++j) {
        int idx = tid + j * 256;
        int q = idx >> 7, d = idx & 127;
        float o = 0.f;
        #pragma unroll
        for (int v = 0; v < 16; ++v) o += sc[q][v] * pvs[v][d];
        int qy = q >> 2, qx = q & 3;
        float cv = cb[d];
        #pragma unroll
        for (int dy = -1; dy <= 1; ++dy)
            #pragma unroll
            for (int dx = -1; dx <= 1; ++dx) {
                int yy = qy + dy, xx = qx + dx;
                if (yy >= 0 && yy < 4 && xx >= 0 && xx < 4)
                    cv += cw[d * 9 + (dy + 1) * 3 + (dx + 1)] * pvs[yy * 4 + xx][d];
            }
        out[(trow0 + q) * 1024 + h * 128 + d] = f2bf(o + cv);
    }
}

// ---------------- window attention (diag/line logic) + LePE conv -> WFm, dsc ----------------
__global__ __launch_bounds__(256) void window_attn_kernel(const float* __restrict__ QKw,
                                                          const float* __restrict__ WV,
                                                          const float* __restrict__ cw,
                                                          const float* __restrict__ cb,
                                                          __bf16* __restrict__ outWFm,
                                                          float* __restrict__ dsc) {
    const int b = blockIdx.x;
    const int tid = threadIdx.x;
    __shared__ float sc[16][16];
    __shared__ float nds[16][16];
    __shared__ float wvs[16][128];

    for (int h = 0; h < 8; ++h) {
        #pragma unroll
        for (int j = 0; j < 8; ++j) {
            int idx = tid + j * 256;
            int vi = idx >> 7, d = idx & 127;
            wvs[vi][d] = WV[((size_t)b * 16 + vi) * 1024 + h * 128 + d];
        }
        const int qi = tid >> 4, ki = tid & 15;
        {
            const float4* qp = (const float4*)&QKw[((size_t)b * 16 + qi) * 2048 + h * 128];
            const float4* kp = (const float4*)&QKw[((size_t)b * 16 + ki) * 2048 + 1024 + h * 128];
            float dot = 0.f;
            #pragma unroll 8
            for (int d4 = 0; d4 < 32; ++d4) {
                float4 a = qp[d4], bb = kp[d4];
                dot += a.x * bb.x + a.y * bb.y + a.z * bb.z + a.w * bb.w;
            }
            sc[qi][ki] = dot * INV_SQRT_D;
        }
        __syncthreads();
        if (tid < 16) {
            float mx = -1e30f;
            #pragma unroll
            for (int k = 0; k < 16; ++k) mx = fmaxf(mx, sc[tid][k]);
            float sm = 0.f;
            #pragma unroll
            for (int k = 0; k < 16; ++k) { float e = __expf(sc[tid][k] - mx); sc[tid][k] = e; sm += e; }
            float inv = 1.f / sm;
            #pragma unroll
            for (int k = 0; k < 16; ++k) sc[tid][k] *= inv;
            float dval = sc[tid][tid];
            bool line = dval < 0.5f;
            dsc[((size_t)b * 8 + h) * 16 + tid] = line ? 0.5f : dval;
            float nscale = line ? (0.5f / (1.0f - dval)) : 1.0f;
            #pragma unroll
            for (int v = 0; v < 16; ++v)
                nds[tid][v] = (v == tid) ? 0.f : nscale * sc[tid][v];
        }
        __syncthreads();
        #pragma unroll
        for (int j = 0; j < 8; ++j) {
            int idx = tid + j * 256;
            int n = idx >> 7, d = idx & 127;
            float o = 0.f;
            #pragma unroll
            for (int v = 0; v < 16; ++v) o += nds[n][v] * wvs[v][d];
            int ny = n >> 2, nx = n & 3;
            float cv = cb[d];
            #pragma unroll
            for (int dy = -1; dy <= 1; ++dy)
                #pragma unroll
                for (int dx = -1; dx <= 1; ++dx) {
                    int yy = ny + dy, xx = nx + dx;
                    if (yy >= 0 && yy < 4 && xx >= 0 && xx < 4)
                        cv += cw[d * 9 + (dy + 1) * 3 + (dx + 1)] * wvs[yy * 4 + xx][d];
                }
            outWFm[((size_t)b * 16 + n) * 1024 + h * 128 + d] = f2bf(o + cv);
        }
        __syncthreads();
    }
}

// ---------------- final: out = x + diag_scaled*pf + wf (scatter back to NCHW) ----------------
__global__ __launch_bounds__(256) void final_kernel(const float* __restrict__ x,
                                                    const float* __restrict__ PF,
                                                    const float* __restrict__ WF,
                                                    const float* __restrict__ dsc,
                                                    float* __restrict__ out) {
    size_t i = (size_t)blockIdx.x * 256 + threadIdx.x;   // over 64*16*128*128
    int X = i & 127, Y = (i >> 7) & 127, c = (i >> 14) & 15, b = (int)(i >> 18);
    int wy = Y >> 5, sy = Y & 31, wx = X >> 5, sx = X & 31;
    int n = wy * 4 + wx;
    int j = (sy * 32 + sx) * 16 + c;
    int h = j >> 11;
    int pat = j >> 10, pdidx = j & 1023;
    float pfv = PF[((size_t)(b * 16 + n) * 16 + pat) * 1024 + pdidx];
    float wfv = WF[((size_t)b * 16 + n) * 16384 + j];
    float dv  = dsc[((size_t)b * 8 + h) * 16 + n];
    out[i] = x[i] + dv * pfv + wfv;
}

// ---------------- launch ----------------
extern "C" void kernel_launch(void* const* d_in, const int* in_sizes, int n_in,
                              void* d_out, int out_size, void* d_ws, size_t ws_size,
                              hipStream_t stream) {
    (void)in_sizes; (void)n_in; (void)out_size; (void)ws_size;
    const float* x        = (const float*)d_in[0];
    const float* pn_g     = (const float*)d_in[1];
    const float* pn_b     = (const float*)d_in[2];
    const float* Wqk_p    = (const float*)d_in[3];
    const float* Wv_p     = (const float*)d_in[4];
    const float* conv_p_w = (const float*)d_in[5];
    const float* conv_p_b = (const float*)d_in[6];
    const float* Wout_p   = (const float*)d_in[7];
    const float* pln_g    = (const float*)d_in[8];
    const float* pln_b    = (const float*)d_in[9];
    const float* Wqk_w    = (const float*)d_in[10];
    const float* wn_g     = (const float*)d_in[11];
    const float* wn_b     = (const float*)d_in[12];
    const float* Wv_w     = (const float*)d_in[13];
    const float* conv_w_w = (const float*)d_in[14];
    const float* conv_w_b = (const float*)d_in[15];
    const float* Wout_w   = (const float*)d_in[16];

    char* w = (char*)d_ws;
    size_t off = 0;
    auto alloc = [&](size_t bytes) -> void* {
        void* p = w + off;
        off += (bytes + 255) & ~(size_t)255;
        return p;
    };

    // transposed bf16 weights: Wt[N][K]
    __bf16* Wqk_p_t  = (__bf16*)alloc(1024LL * 2048 * 2);
    __bf16* Wv_p_t   = (__bf16*)alloc(1024LL * 1024 * 2);
    __bf16* Wout_p_t = (__bf16*)alloc(1024LL * 1024 * 2);
    __bf16* Wqk_w_t  = (__bf16*)alloc(1024LL * 2048 * 2);
    __bf16* Wv_w_t   = (__bf16*)alloc(16384LL * 1024 * 2);
    __bf16* Wout_w_t = (__bf16*)alloc(1024LL * 16384 * 2);

    __bf16* FLbf = (__bf16*)alloc(16384LL * 1024 * 2);
    float*  QK   = (float*) alloc(16384LL * 2048 * 4);   // reused as PF later
    float*  PV   = (float*) alloc(16384LL * 1024 * 4);   // reused as WF later
    __bf16* PFm  = (__bf16*)alloc(16384LL * 1024 * 2);
    __bf16* PLn  = (__bf16*)alloc(1024LL * 1024 * 2);
    float*  QKw  = (float*) alloc(1024LL * 2048 * 4);
    __bf16* XWn  = (__bf16*)alloc(1024LL * 16384 * 2);
    float*  WV   = (float*) alloc(1024LL * 1024 * 4);
    __bf16* WFm  = (__bf16*)alloc(1024LL * 1024 * 2);
    float*  dsc  = (float*) alloc(64LL * 8 * 16 * 4);
    float*  PF   = QK;   // alias: QK dead after patch_attn
    float*  WF   = PV;   // alias: PV dead after patch_attn

    auto cvtT = [&](const float* src, __bf16* dst, int K, int N) {
        cvt_transpose_kernel<<<dim3(N / 32, K / 32), 256, 0, stream>>>(src, dst, K, N);
    };
    cvtT(Wqk_p, Wqk_p_t, 1024, 2048);
    cvtT(Wv_p, Wv_p_t, 1024, 1024);
    cvtT(Wout_p, Wout_p_t, 1024, 1024);
    cvtT(Wqk_w, Wqk_w_t, 1024, 2048);
    cvtT(Wv_w, Wv_w_t, 16384, 1024);
    cvtT(Wout_w, Wout_w_t, 1024, 16384);

    // ---- patch path ----
    patch_ln_kernel<<<16384, 256, 0, stream>>>(x, pn_g, pn_b, FLbf);
    gemm_bf16_kernel<<<dim3(2048 / 128, 16384 / 128), 256, 0, stream>>>(
        FLbf, Wqk_p_t, QK, 16384, 2048, 1024);
    gemm_bf16_kernel<<<dim3(1024 / 128, 16384 / 128), 256, 0, stream>>>(
        FLbf, Wv_p_t, PV, 16384, 1024, 1024);
    patch_attn_kernel<<<64 * 16 * 8, 256, 0, stream>>>(QK, PV, conv_p_w, conv_p_b, PFm);
    gemm_bf16_kernel<<<dim3(1024 / 128, 16384 / 128), 256, 0, stream>>>(
        PFm, Wout_p_t, PF, 16384, 1024, 1024);

    // ---- window path ----
    pool_ln_kernel<<<1024, 256, 0, stream>>>(x, pln_g, pln_b, PLn);
    gemm_bf16_kernel<<<dim3(2048 / 128, 1024 / 128), 256, 0, stream>>>(
        PLn, Wqk_w_t, QKw, 1024, 2048, 1024);
    xw_ln_kernel<<<1024, 256, 0, stream>>>(x, wn_g, wn_b, XWn);
    gemm_bf16_kernel<<<dim3(1024 / 128, 1024 / 128), 256, 0, stream>>>(
        XWn, Wv_w_t, WV, 1024, 1024, 16384);
    window_attn_kernel<<<64, 256, 0, stream>>>(QKw, WV, conv_w_w, conv_w_b, WFm, dsc);
    gemm_bf16_kernel<<<dim3(16384 / 128, 1024 / 128), 256, 0, stream>>>(
        WFm, Wout_w_t, WF, 1024, 16384, 1024);

    // ---- final combine ----
    final_kernel<<<(64LL * 16 * 128 * 128) / 256, 256, 0, stream>>>(x, PF, WF, dsc, (float*)d_out);
}
